// AGILE_7696581394505
// MI455X (gfx1250) — matrix operand order repaired and checked by reading, not compiled, not run on hardware
//
#include <hip/hip_runtime.h>
#include <hip/hip_bf16.h>
#include <math.h>

// ---------------------------------------------------------------------------
// Problem constants (match reference)
// ---------------------------------------------------------------------------
#define NN      100000
#define EE      200000
#define BB      4096
#define DD      300
#define FEAT    512
#define NLAYER  5
#define LD_H    320          // padded row stride for D=300 fp32 buffers
#define LD_MID  640          // padded row stride for 2D=600 fp32 buffer
#define BN_EPS  1e-5f

typedef __attribute__((ext_vector_type(16))) __bf16 v16bf;
typedef __attribute__((ext_vector_type(8)))  float  v8f;

// ---------------------------------------------------------------------------
// Fragment loaders (CDNA5 WMMA 16x16x32 bf16 VGPR layouts, wave32)
// A 16x32: lane row = lane%16; half=lane>=16; elements 0..7 <- K = half*8+0..7,
//          elements 8..15 <- K = half*8+16..23  (two contiguous 16B chunks)
// B 32x16 (from transposed weights [Np,Kp]): lane col = lane%16;
//          elements 0..7 <- K = half*8+0..7, elements 8..15 <- K = half*8+16..23
// ---------------------------------------------------------------------------
__device__ __forceinline__ v16bf load_afrag(const __bf16* p) {
  v16bf v;
  ((uint4*)&v)[0] = *(const uint4*)(p);
  ((uint4*)&v)[1] = *(const uint4*)(p + 16);
  return v;
}
__device__ __forceinline__ v16bf load_bfrag(const __bf16* p) {
  v16bf v;
  ((uint4*)&v)[0] = *(const uint4*)(p);
  ((uint4*)&v)[1] = *(const uint4*)(p + 16);
  return v;
}

__device__ __forceinline__ void epilogue_store(float* crow, int col, int nlog,
                                               const float* bias, int act, float v) {
  if (col < nlog) {
    v += bias[col];
    if (act == 1) v = fmaxf(v, 0.f);
    else if (act == 2) v = (v > 20.f) ? v : log1pf(expf(v));
    crow[col] = v;
  }
}

// ---------------------------------------------------------------------------
// Tiled bf16 WMMA GEMM: C[M,Nlog] = A[M,Kp] * Bt^T + bias, optional act.
//   A  : bf16 [M, Kp], Kp % 32 == 0, M % 16 == 0
//   Bt : bf16 [Np, Kp] (weights pre-transposed, zero padded), Np % 64 == 0
//   C  : fp32, row stride ldc; only cols < nlog are written
// Block = 128 threads (4 waves); wave computes a 16x64 slab (4 WMMA tiles).
// ---------------------------------------------------------------------------
__global__ __launch_bounds__(128)
void gemm_bf16_wmma(const __bf16* __restrict__ A, const __bf16* __restrict__ Bt,
                    const float* __restrict__ bias, float* __restrict__ C,
                    int M, int Kp, int nlog, int ldc, int act) {
  const int lane   = threadIdx.x & 31;
  const int wave   = threadIdx.x >> 5;
  const int half   = lane >> 4;
  const int lanelo = lane & 15;
  const int strip  = blockIdx.x * 4 + wave;     // 16-row strip index
  const int row0   = strip * 16;
  if (row0 >= M) return;                        // uniform across the wave
  const int cb0 = blockIdx.y * 64;

  v8f acc0 = {}, acc1 = {}, acc2 = {}, acc3 = {};
  const __bf16* aptr  = A  + (size_t)(row0 + lanelo) * Kp + half * 8;
  const __bf16* bptr  = Bt + (size_t)(cb0 + lanelo) * Kp + half * 8;
  const size_t  bstep = (size_t)16 * Kp;        // next 16-column tile

  for (int k = 0; k < Kp; k += 32) {
    v16bf a  = load_afrag(aptr + k);
    v16bf b0 = load_bfrag(bptr + k);
    v16bf b1 = load_bfrag(bptr + bstep     + k);
    v16bf b2 = load_bfrag(bptr + bstep * 2 + k);
    v16bf b3 = load_bfrag(bptr + bstep * 3 + k);
    acc0 = __builtin_amdgcn_wmma_f32_16x16x32_bf16(false, a, false, b0, (short)0, acc0, false, false);
    acc1 = __builtin_amdgcn_wmma_f32_16x16x32_bf16(false, a, false, b1, (short)0, acc1, false, false);
    acc2 = __builtin_amdgcn_wmma_f32_16x16x32_bf16(false, a, false, b2, (short)0, acc2, false, false);
    acc3 = __builtin_amdgcn_wmma_f32_16x16x32_bf16(false, a, false, b3, (short)0, acc3, false, false);
  }

  // D layout: element r -> row = r + 8*half, col = cb0 + lanelo (+16 per tile)
  const int rbase = row0 + half * 8;
  #pragma unroll
  for (int r = 0; r < 8; ++r) {
    float* crow = C + (size_t)(rbase + r) * ldc;
    const int c = cb0 + lanelo;
    epilogue_store(crow, c,      nlog, bias, act, acc0[r]);
    epilogue_store(crow, c + 16, nlog, bias, act, acc1[r]);
    epilogue_store(crow, c + 32, nlog, bias, act, acc2[r]);
    epilogue_store(crow, c + 48, nlog, bias, act, acc3[r]);
  }
}

// ---------------------------------------------------------------------------
// Elementwise / packing / graph kernels
// ---------------------------------------------------------------------------
__global__ void k_zero(float* p, int n) {
  int i = blockIdx.x * blockDim.x + threadIdx.x;
  if (i < n) p[i] = 0.f;
}

// fp32 [M, src_ld] (first K cols) -> bf16 [M, Kp] zero-padded
__global__ void k_pack_bf16(const float* __restrict__ src, __bf16* __restrict__ dst,
                            int M, int src_ld, int K, int Kp) {
  int i = blockIdx.x * blockDim.x + threadIdx.x;
  if (i >= M * Kp) return;
  int m = i / Kp, k = i - m * Kp;
  dst[i] = (k < K) ? (__bf16)src[(size_t)m * src_ld + k] : (__bf16)0.f;
}

// fp32 weights [K, N] row-major -> bf16 transposed [Np, Kp] zero-padded
__global__ void k_wpack_bf16(const float* __restrict__ w, __bf16* __restrict__ dst,
                             int K, int N, int Kp, int Np) {
  int i = blockIdx.x * blockDim.x + threadIdx.x;
  if (i >= Np * Kp) return;
  int n = i / Kp, k = i - n * Kp;
  dst[i] = (n < N && k < K) ? (__bf16)w[(size_t)k * N + n] : (__bf16)0.f;
}

// h = x_emb1[x[:,0]] + x_emb2[x[:,1]]
__global__ void k_embed(const int* __restrict__ x,
                        const float* __restrict__ e1, const float* __restrict__ e2,
                        float* __restrict__ h) {
  int n = blockIdx.x, d = threadIdx.x;
  if (d >= DD) return;
  int a0 = x[2 * n], a1 = x[2 * n + 1];
  h[(size_t)n * LD_H + d] = e1[(size_t)a0 * DD + d] + e2[(size_t)a1 * DD + d];
}

// aggr[dst] += h[src] + edge_emb1[bond] + edge_emb2[dir]  (E edges + N self loops)
__global__ void k_scatter(const float* __restrict__ h,
                          const int* __restrict__ ei, const int* __restrict__ eatt,
                          const float* __restrict__ ee1, const float* __restrict__ ee2,
                          float* __restrict__ aggr) {
  int m = blockIdx.x, d = threadIdx.x;
  if (d >= DD) return;
  int s, t, bond, dir;
  if (m < EE) {
    s = ei[m]; t = ei[EE + m];
    bond = eatt[2 * m]; dir = eatt[2 * m + 1];
  } else {
    s = t = m - EE; bond = 4; dir = 0;
  }
  float msg = h[(size_t)s * LD_H + d] + ee1[bond * DD + d] + ee2[dir * DD + d];
  atomicAdd(&aggr[(size_t)t * LD_H + d], msg);
}

// per-column mean / rstd over M rows (biased variance, matches reference)
__global__ void k_bn_stats(const float* __restrict__ h, int M,
                           float* __restrict__ mean, float* __restrict__ rstd) {
  int col = blockIdx.x;
  __shared__ float s1[256], s2[256];
  float a = 0.f, b = 0.f;
  for (int r = threadIdx.x; r < M; r += 256) {
    float v = h[(size_t)r * LD_H + col];
    a += v; b += v * v;
  }
  s1[threadIdx.x] = a; s2[threadIdx.x] = b;
  __syncthreads();
  for (int s = 128; s > 0; s >>= 1) {
    if (threadIdx.x < s) { s1[threadIdx.x] += s1[threadIdx.x + s];
                           s2[threadIdx.x] += s2[threadIdx.x + s]; }
    __syncthreads();
  }
  if (threadIdx.x == 0) {
    float mu  = s1[0] / (float)M;
    float var = s2[0] / (float)M - mu * mu;
    mean[col] = mu;
    rstd[col] = rsqrtf(var + BN_EPS);
  }
}

__global__ void k_bn_apply(float* __restrict__ h,
                           const float* __restrict__ mean, const float* __restrict__ rstd,
                           const float* __restrict__ gamma, const float* __restrict__ beta,
                           int do_relu) {
  int n = blockIdx.x, d = threadIdx.x;
  if (d >= DD) return;
  size_t i = (size_t)n * LD_H + d;
  float v = (h[i] - mean[d]) * rstd[d] * gamma[d] + beta[d];
  if (do_relu) v = fmaxf(v, 0.f);
  h[i] = v;
}

__global__ void k_pool_accum(const float* __restrict__ h, const int* __restrict__ batch,
                             float* __restrict__ pooled, float* __restrict__ counts) {
  int n = blockIdx.x, d = threadIdx.x;
  int bt = batch[n];
  if (d < DD)        atomicAdd(&pooled[(size_t)bt * LD_H + d], h[(size_t)n * LD_H + d]);
  else if (d == 319) atomicAdd(&counts[bt], 1.f);
}

__global__ void k_pool_div(float* __restrict__ pooled, const float* __restrict__ counts) {
  int b = blockIdx.x, d = threadIdx.x;
  if (d >= DD) return;
  pooled[(size_t)b * LD_H + d] /= fmaxf(counts[b], 1.f);
}

// pred[b, o] = sum_k p2[b,k] * Wp3[k,o] + bp3[o]   (N=2 tail, plain VALU)
__global__ void k_pred(const float* __restrict__ p2, const float* __restrict__ Wp3,
                       const float* __restrict__ bp3, float* __restrict__ pred) {
  int i = blockIdx.x * blockDim.x + threadIdx.x;
  if (i >= BB * 2) return;
  int b = i >> 1, o = i & 1;
  float s = bp3[o];
  const float* row = p2 + (size_t)b * 256;
  for (int k = 0; k < 256; ++k) s += row[k] * Wp3[2 * k + o];
  pred[i] = s;
}

// ---------------------------------------------------------------------------
// Host orchestration
// ---------------------------------------------------------------------------
static inline int cdiv(int a, int b) { return (a + b - 1) / b; }

extern "C" void kernel_launch(void* const* d_in, const int* in_sizes, int n_in,
                              void* d_out, int out_size, void* d_ws, size_t ws_size,
                              hipStream_t stream) {
  (void)in_sizes; (void)n_in; (void)out_size; (void)ws_size;

  const int*   x      = (const int*)  d_in[0];
  const int*   ei     = (const int*)  d_in[1];
  const int*   eatt   = (const int*)  d_in[2];
  const int*   batch  = (const int*)  d_in[3];
  const float* x_emb1 = (const float*)d_in[4];
  const float* x_emb2 = (const float*)d_in[5];
  const float* eemb1  = (const float*)d_in[6];
  const float* eemb2  = (const float*)d_in[7];
  const float* W1     = (const float*)d_in[8];
  const float* b1     = (const float*)d_in[9];
  const float* W2     = (const float*)d_in[10];
  const float* b2     = (const float*)d_in[11];
  const float* gamma  = (const float*)d_in[12];
  const float* beta   = (const float*)d_in[13];
  const float* W_feat = (const float*)d_in[14];
  const float* b_feat = (const float*)d_in[15];
  const float* Wp1    = (const float*)d_in[16];
  const float* bp1    = (const float*)d_in[17];
  const float* Wp2    = (const float*)d_in[18];
  const float* bp2    = (const float*)d_in[19];
  const float* Wp3    = (const float*)d_in[20];
  const float* bp3    = (const float*)d_in[21];

  // workspace carve-out
  char* ws = (char*)d_ws;
  size_t off = 0;
  auto carve = [&](size_t bytes) -> char* {
    char* p = ws + off;
    off = (off + bytes + 255) & ~(size_t)255;
    return p;
  };
  float*  hbuf   = (float*)carve((size_t)NN * LD_H  * 4);
  float*  aggr   = (float*)carve((size_t)NN * LD_H  * 4);
  float*  mid    = (float*)carve((size_t)NN * LD_MID * 4);
  __bf16* A16    = (__bf16*)carve((size_t)NN * LD_MID * 2);
  __bf16* Wt16   = (__bf16*)carve((size_t)640 * 640 * 2);
  float*  meanb  = (float*)carve(320 * 4);
  float*  rstdb  = (float*)carve(320 * 4);
  float*  pooled = (float*)carve((size_t)BB * LD_H * 4);
  float*  counts = (float*)carve((size_t)BB * 4);
  __bf16* P16    = (__bf16*)carve((size_t)BB * 640 * 2);
  float*  p1buf  = (float*)carve((size_t)BB * 256 * 4);
  float*  p2buf  = (float*)carve((size_t)BB * 256 * 4);

  float* hf   = (float*)d_out;              // [B, FEAT]
  float* pred = hf + (size_t)BB * FEAT;     // [B, 2]

  // node embeddings
  k_embed<<<NN, 320, 0, stream>>>(x, x_emb1, x_emb2, hbuf);

  // GNN layers
  for (int l = 0; l < NLAYER; ++l) {
    // aggr = segment_sum(h[src] + edge_emb)
    k_zero<<<cdiv(NN * LD_H, 256), 256, 0, stream>>>(aggr, NN * LD_H);
    k_scatter<<<EE + NN, 320, 0, stream>>>(hbuf, ei, eatt,
                                           eemb1 + (size_t)l * 5 * DD,
                                           eemb2 + (size_t)l * 3 * DD, aggr);

    // mid = relu(aggr @ W1[l] + b1[l])   [N, 600] via bf16 WMMA (Kp=320, Np=640)
    k_pack_bf16<<<cdiv(NN * 320, 256), 256, 0, stream>>>(aggr, A16, NN, LD_H, DD, 320);
    k_wpack_bf16<<<cdiv(640 * 320, 256), 256, 0, stream>>>(W1 + (size_t)l * DD * 2 * DD,
                                                           Wt16, DD, 2 * DD, 320, 640);
    {
      dim3 g(cdiv(NN / 16, 4), 640 / 64);
      gemm_bf16_wmma<<<g, 128, 0, stream>>>(A16, Wt16, b1 + (size_t)l * 2 * DD,
                                            mid, NN, 320, 2 * DD, LD_MID, /*relu*/1);
    }

    // h = mid @ W2[l] + b2[l]   [N, 300] via bf16 WMMA (Kp=640, Np=320)
    k_pack_bf16<<<cdiv(NN * 640, 256), 256, 0, stream>>>(mid, A16, NN, LD_MID, 2 * DD, 640);
    k_wpack_bf16<<<cdiv(320 * 640, 256), 256, 0, stream>>>(W2 + (size_t)l * 2 * DD * DD,
                                                           Wt16, 2 * DD, DD, 640, 320);
    {
      dim3 g(cdiv(NN / 16, 4), 320 / 64);
      gemm_bf16_wmma<<<g, 128, 0, stream>>>(A16, Wt16, b2 + (size_t)l * DD,
                                            hbuf, NN, 640, DD, LD_H, /*none*/0);
    }

    // batch norm (training-mode biased var) + relu (except last layer)
    k_bn_stats<<<DD, 256, 0, stream>>>(hbuf, NN, meanb, rstdb);
    k_bn_apply<<<NN, 320, 0, stream>>>(hbuf, meanb, rstdb,
                                       gamma + (size_t)l * DD, beta + (size_t)l * DD,
                                       (l < NLAYER - 1) ? 1 : 0);
  }

  // global mean pooling per graph
  k_zero<<<cdiv(BB * LD_H, 256), 256, 0, stream>>>(pooled, BB * LD_H);
  k_zero<<<cdiv(BB, 256), 256, 0, stream>>>(counts, BB);
  k_pool_accum<<<NN, 320, 0, stream>>>(hbuf, batch, pooled, counts);
  k_pool_div<<<BB, 320, 0, stream>>>(pooled, counts);

  // hf = pooled @ W_feat + b_feat   [B, 512]  (Kp=320, Np=512)
  k_pack_bf16<<<cdiv(BB * 320, 256), 256, 0, stream>>>(pooled, P16, BB, LD_H, DD, 320);
  k_wpack_bf16<<<cdiv(512 * 320, 256), 256, 0, stream>>>(W_feat, Wt16, DD, FEAT, 320, 512);
  {
    dim3 g(cdiv(BB / 16, 4), 512 / 64);
    gemm_bf16_wmma<<<g, 128, 0, stream>>>(P16, Wt16, b_feat, hf, BB, 320, FEAT, FEAT, 0);
  }

  // p1 = softplus(hf @ Wp1 + bp1)   [B, 256]  (Kp=512, Np=256)
  k_pack_bf16<<<cdiv(BB * 512, 256), 256, 0, stream>>>(hf, P16, BB, FEAT, FEAT, 512);
  k_wpack_bf16<<<cdiv(256 * 512, 256), 256, 0, stream>>>(Wp1, Wt16, FEAT, 256, 512, 256);
  {
    dim3 g(cdiv(BB / 16, 4), 256 / 64);
    gemm_bf16_wmma<<<g, 128, 0, stream>>>(P16, Wt16, bp1, p1buf, BB, 512, 256, 256, /*softplus*/2);
  }

  // p2 = softplus(p1 @ Wp2 + bp2)   [B, 256]  (Kp=256, Np=256)
  k_pack_bf16<<<cdiv(BB * 256, 256), 256, 0, stream>>>(p1buf, P16, BB, 256, 256, 256);
  k_wpack_bf16<<<cdiv(256 * 256, 256), 256, 0, stream>>>(Wp2, Wt16, 256, 256, 256, 256);
  {
    dim3 g(cdiv(BB / 16, 4), 256 / 64);
    gemm_bf16_wmma<<<g, 128, 0, stream>>>(P16, Wt16, bp2, p2buf, BB, 256, 256, 256, /*softplus*/2);
  }

  // pred = p2 @ Wp3 + bp3   [B, 2]
  k_pred<<<cdiv(BB * 2, 256), 256, 0, stream>>>(p2buf, Wp3, bp3, pred);
}
